// CMPNEncoder_84920093377278
// MI455X (gfx1250) — matrix operand order repaired and checked by reading, not compiled, not run on hardware
//
#include <hip/hip_runtime.h>
#include <math.h>

typedef __bf16 bf16;
typedef __attribute__((ext_vector_type(16))) __bf16 v16bf;
typedef __attribute__((ext_vector_type(8)))  float   v8f;
typedef unsigned int u32x4 __attribute__((ext_vector_type(4)));

union Frag { u32x4 u[2]; v16bf v; };

#define TPB 256

// Problem constants (fixed by setup_inputs)
#define HDIM   300
#define APM    64
#define NMOL   512
#define NATOMS 32769           // 1 + 512*64
#define NBONDS 131073          // 1 + 512*64*4
#define MA     33024           // NATOMS padded to 256
#define MB     131328          // NBONDS padded to 256
#define MX     32768           // 512*64 (already multiple of 256)
#define KAT    160             // 133 -> pad 32
#define KBD    160             // 147 -> pad 32
#define KH     320             // 300 -> pad 32
#define KC     928             // 900 -> pad 32
#define KO     608             // 600 -> pad 32
#define NPH    320             // 300 -> pad 64 (N tiles)
#define NPG    960             // 900 -> pad 64

// ---------------------------------------------------------------------------
// Tiled WMMA GEMM (bf16 x bf16 -> f32):
//   C[M,N] = epilogue( A[Mpad,Kp] @ B[Kp,N] )  with B stored transposed as
//   Bt[Npad,Kp] (row n = column n of B). Block tile 256x64, 8 waves, each
//   wave computes a 32-row strip (2 A frags) x four 16x16 WMMA tiles sharing
//   the B fragments -> 8 WMMAs per 12 ds_load_b128 per K-chunk.
// mode bits: 1 = relu, 2 = +bias[col], 4 = +resid[row*ldc+col]
// ---------------------------------------------------------------------------
__global__ __launch_bounds__(256)
void k_wmma_gemm(const bf16* __restrict__ A, const bf16* __restrict__ Bt,
                 const float* __restrict__ bias, const float* __restrict__ resid,
                 float* __restrict__ C, int M, int N, int Kp, int ldc, int mode)
{
  __shared__ __align__(16) bf16 As[256 * 40];   // 32 K-cols + 8 pad per row
  __shared__ __align__(16) bf16 Bs[64 * 40];
  const int tid  = threadIdx.x;
  const int wave = tid >> 5;
  const int lane = tid & 31;
  const int half = lane >> 4;
  const int mn   = lane & 15;
  const long m0  = (long)blockIdx.x * 256;
  const long n0  = (long)blockIdx.y * 64;
  const int  kq  = Kp >> 3;                     // 16B chunks per row

  v8f acc[2][4];
  #pragma unroll
  for (int i = 0; i < 2; ++i)
    #pragma unroll
    for (int j = 0; j < 4; ++j) acc[i][j] = 0.0f;

  const int nk = Kp >> 5;
  for (int kb = 0; kb < nk; ++kb) {
    // ---- stage A (256x32) and Bt (64x32) tiles into LDS, coalesced b128 ----
    const u32x4* ag = (const u32x4*)(A  + m0 * (long)Kp + (long)kb * 32);
    const u32x4* bg = (const u32x4*)(Bt + n0 * (long)Kp + (long)kb * 32);
    {
      #pragma unroll
      for (int i = 0; i < 4; ++i) {
        const int c = tid + i * 256;
        u32x4 d = ag[(long)(c >> 2) * kq + (c & 3)];
        *(u32x4*)&As[(c >> 2) * 40 + (c & 3) * 8] = d;
      }
      u32x4 db = bg[(long)(tid >> 2) * kq + (tid & 3)];
      *(u32x4*)&Bs[(tid >> 2) * 40 + (tid & 3) * 8] = db;
    }
    __syncthreads();

    // ---- A fragments: lane holds rows (wave*32+mn) and (wave*32+16+mn);
    //      ISA A layout: elems 0..7  -> K = half*8 + 0..7
    //                    elems 8..15 -> K = 16 + half*8 + 0..7
    Frag a0, a1;
    const bf16* ap0 = &As[(wave * 32 + mn) * 40 + half * 8];
    a0.u[0] = *(const u32x4*)ap0;
    a0.u[1] = *(const u32x4*)(ap0 + 16);
    const bf16* ap1 = ap0 + 16 * 40;
    a1.u[0] = *(const u32x4*)ap1;
    a1.u[1] = *(const u32x4*)(ap1 + 16);

    #pragma unroll
    for (int j = 0; j < 4; ++j) {
      // ---- B fragment: lane holds col (j*16+mn); K = half*8 + 0..7, 16 + half*8 + 0..7
      Frag b;
      const bf16* bp = &Bs[(j * 16 + mn) * 40 + half * 8];
      b.u[0] = *(const u32x4*)bp;
      b.u[1] = *(const u32x4*)(bp + 16);
      acc[0][j] = __builtin_amdgcn_wmma_f32_16x16x32_bf16(
          false, a0.v, false, b.v, (short)0, acc[0][j], false, false);
      acc[1][j] = __builtin_amdgcn_wmma_f32_16x16x32_bf16(
          false, a1.v, false, b.v, (short)0, acc[1][j], false, false);
    }
    __syncthreads();
  }

  // ---- epilogue: C/D layout VGPR r -> M = r + half*8, lane -> N = mn ----
  #pragma unroll
  for (int i = 0; i < 2; ++i) {
    #pragma unroll
    for (int j = 0; j < 4; ++j) {
      const long col = n0 + j * 16 + mn;
      if (col >= N) continue;
      const float bs = (mode & 2) ? bias[col] : 0.0f;
      #pragma unroll
      for (int r = 0; r < 8; ++r) {
        const long row = m0 + wave * 32 + i * 16 + half * 8 + r;
        if (row >= M) continue;
        float v = acc[i][j][r] + bs;
        if (mode & 4) v += resid[row * (long)ldc + col];
        if (mode & 1) v = fmaxf(v, 0.0f);
        C[row * (long)ldc + col] = v;
      }
    }
  }
}

// ---------------------------------------------------------------------------
// Elementwise / data-movement kernels
// ---------------------------------------------------------------------------
__global__ void k_zero_bf(bf16* p, long n)
{
  long i = (long)blockIdx.x * blockDim.x + threadIdx.x;
  long base = i * 8;
  if (base + 8 <= n) { u32x4 z = 0; *(u32x4*)(p + base) = z; }
  else { for (long j = base; j < n; ++j) p[j] = (bf16)0.0f; }
}

// f32 [rows,cols] -> bf16 [Mpad,Kp], zero padded
__global__ void k_conv_act(const float* __restrict__ src, bf16* __restrict__ dst,
                           int rows, int cols, int Kp, long n)
{
  long idx = (long)blockIdx.x * blockDim.x + threadIdx.x;
  if (idx >= n) return;
  int r = (int)(idx / Kp), c = (int)(idx % Kp);
  float v = (r < rows && c < cols) ? src[(long)r * cols + c] : 0.0f;
  dst[idx] = (bf16)v;
}

// weight f32 -> bf16 transposed store dst[Np,Kp]
// mode 0: B = src[K,N]        -> dst[n*Kp+k] = src[k*N+n]
// mode 1: B = src^T, src[N,K] -> dst[n*Kp+k] = src[n*K+k]
__global__ void k_conv_w(const float* __restrict__ src, bf16* __restrict__ dst,
                         int K, int N, int Kp, int mode, long n)
{
  long idx = (long)blockIdx.x * blockDim.x + threadIdx.x;
  if (idx >= n) return;
  int ncol = (int)(idx / Kp), k = (int)(idx % Kp);
  float v = 0.0f;
  if (k < K && ncol < N)
    v = (mode == 0) ? src[(long)k * N + ncol] : src[(long)ncol * K + k];
  dst[idx] = (bf16)v;
}

// agg[v] = sum_i(mb[a2b[v,i]]) * max_i(mb[a2b[v,i]]);  out = base + agg
__global__ void k_aggregate(const float* __restrict__ mb, const int* __restrict__ a2b,
                            const float* __restrict__ base, float* __restrict__ out)
{
  long idx = (long)blockIdx.x * blockDim.x + threadIdx.x;
  if (idx >= (long)NATOMS * HDIM) return;
  int v = (int)(idx / HDIM), h = (int)(idx % HDIM);
  float s = 0.0f, mx = -3.4e38f;
  #pragma unroll
  for (int i = 0; i < 6; ++i) {
    int b = a2b[(long)v * 6 + i];
    float val = mb[(long)b * HDIM + h];
    s += val;
    mx = fmaxf(mx, val);
  }
  float agg = s * mx;
  out[idx] = (base ? base[idx] : 0.0f) + agg;
}

// mb_bf[e] = bf16( matom[b2a[e]] - mb[b2revb[e]] ), padded cols stay zero
__global__ void k_bond_pre(const float* __restrict__ matom, const float* __restrict__ mb,
                           const int* __restrict__ b2a, const int* __restrict__ b2revb,
                           bf16* __restrict__ mb_bf)
{
  long idx = (long)blockIdx.x * blockDim.x + threadIdx.x;
  if (idx >= (long)NBONDS * HDIM) return;
  int e = (int)(idx / HDIM), h = (int)(idx % HDIM);
  float v = matom[(long)b2a[e] * HDIM + h] - mb[(long)b2revb[e] * HDIM + h];
  mb_bf[(long)e * KH + h] = (bf16)v;
}

// build bf16 concat [agg3 | matom | input_atom] into [MA, KC]
__global__ void k_concat(const float* __restrict__ agg3, const float* __restrict__ matom,
                         const float* __restrict__ iatom, bf16* __restrict__ dst)
{
  long idx = (long)blockIdx.x * blockDim.x + threadIdx.x;
  if (idx >= (long)NATOMS * HDIM) return;
  int v = (int)(idx / HDIM), h = (int)(idx % HDIM);
  long base = (long)v * KC;
  dst[base + h]           = (bf16)agg3[idx];
  dst[base + HDIM + h]    = (bf16)matom[idx];
  dst[base + 2*HDIM + h]  = (bf16)iatom[idx];
}

// X_bf[r] = bf16(relu(agg2[1+r] + gbias))   r = b*64 + t
__global__ void k_msgx(const float* __restrict__ agg2, const float* __restrict__ gbias,
                       bf16* __restrict__ X)
{
  long idx = (long)blockIdx.x * blockDim.x + threadIdx.x;
  if (idx >= (long)MX * HDIM) return;
  int r = (int)(idx / HDIM), h = (int)(idx % HDIM);
  float v = fmaxf(agg2[(long)(1 + r) * HDIM + h] + gbias[h], 0.0f);
  X[(long)r * KH + h] = (bf16)v;
}

// h0[b] = max_a agg2[1 + b*64 + a]
__global__ void k_h0(const float* __restrict__ agg2, float* __restrict__ h0)
{
  long idx = (long)blockIdx.x * blockDim.x + threadIdx.x;
  if (idx >= (long)NMOL * HDIM) return;
  int b = (int)(idx / HDIM), h = (int)(idx % HDIM);
  float mx = -3.4e38f;
  for (int a = 0; a < APM; ++a)
    mx = fmaxf(mx, agg2[(long)(1 + b * APM + a) * HDIM + h]);
  h0[idx] = mx;
}

__global__ void k_hinit(const float* __restrict__ h0, float* hF, float* hB,
                        bf16* hFbf, bf16* hBbf)
{
  long idx = (long)blockIdx.x * blockDim.x + threadIdx.x;
  if (idx >= (long)NMOL * HDIM) return;
  int b = (int)(idx / HDIM), h = (int)(idx % HDIM);
  float v = h0[idx];
  hF[idx] = v; hB[idx] = v;
  hFbf[(long)b * KH + h] = (bf16)v;
  hBbf[(long)b * KH + h] = (bf16)v;
}

// gru_out row 0 = [msg0 | msg0], msg0 = relu(agg2[0] + gbias)
__global__ void k_row0(const float* __restrict__ agg2, const float* __restrict__ gbias,
                       bf16* __restrict__ gru)
{
  int h = threadIdx.x;
  if (h >= HDIM) return;
  float v = fmaxf(agg2[h] + gbias[h], 0.0f);
  gru[h] = (bf16)v;
  gru[HDIM + h] = (bf16)v;
}

// Fused GRU gates for one timestep (biases already folded into GI / gh GEMMs)
__global__ void k_gate(const float* __restrict__ GI, const float* __restrict__ gh,
                       float* __restrict__ h, bf16* __restrict__ h_bf,
                       bf16* __restrict__ gru, int t, int outOff)
{
  long idx = (long)blockIdx.x * blockDim.x + threadIdx.x;
  if (idx >= (long)NMOL * HDIM) return;
  int b = (int)(idx / HDIM), j = (int)(idx % HDIM);
  long gr = (long)(b * APM + t) * (3 * HDIM);
  long gb = (long)b * (3 * HDIM);
  float gi_r = GI[gr + j], gi_z = GI[gr + HDIM + j], gi_n = GI[gr + 2*HDIM + j];
  float gh_r = gh[gb + j], gh_z = gh[gb + HDIM + j], gh_n = gh[gb + 2*HDIM + j];
  float r = 1.0f / (1.0f + expf(-(gi_r + gh_r)));
  float z = 1.0f / (1.0f + expf(-(gi_z + gh_z)));
  float nn = tanhf(gi_n + r * gh_n);
  float hp = h[idx];
  float hn = (1.0f - z) * nn + z * hp;
  h[idx] = hn;
  h_bf[(long)b * KH + j] = (bf16)hn;
  gru[(long)(1 + b * APM + t) * KO + outOff + j] = (bf16)hn;
}

// mol_vecs[b] = mean_a AH[1 + b*64 + a]
__global__ void k_mean(const float* __restrict__ AH, float* __restrict__ out)
{
  long idx = (long)blockIdx.x * blockDim.x + threadIdx.x;
  if (idx >= (long)NMOL * HDIM) return;
  int b = (int)(idx / HDIM), h = (int)(idx % HDIM);
  float s = 0.0f;
  for (int a = 0; a < APM; ++a)
    s += AH[(long)(1 + b * APM + a) * HDIM + h];
  out[idx] = s * (1.0f / (float)APM);
}

// ---------------------------------------------------------------------------
extern "C" void kernel_launch(void* const* d_in, const int* in_sizes, int n_in,
                              void* d_out, int out_size, void* d_ws, size_t ws_size,
                              hipStream_t stream)
{
  (void)in_sizes; (void)n_in; (void)out_size; (void)ws_size;

  const float* f_atoms  = (const float*)d_in[0];
  const float* f_bonds  = (const float*)d_in[1];
  const float* W_i_atom = (const float*)d_in[2];
  const float* W_i_bond = (const float*)d_in[3];
  const float* W_h_0    = (const float*)d_in[4];
  const float* W_h_1    = (const float*)d_in[5];
  const float* W_lr     = (const float*)d_in[6];
  const float* W_o      = (const float*)d_in[7];
  const float* b_o      = (const float*)d_in[8];
  const float* gru_bias = (const float*)d_in[9];
  const float* Wih_f    = (const float*)d_in[10];
  const float* Whh_f    = (const float*)d_in[11];
  const float* bih_f    = (const float*)d_in[12];
  const float* bhh_f    = (const float*)d_in[13];
  const float* Wih_b    = (const float*)d_in[14];
  const float* Whh_b    = (const float*)d_in[15];
  const float* bih_b    = (const float*)d_in[16];
  const float* bhh_b    = (const float*)d_in[17];
  const int*   a2b      = (const int*)d_in[18];
  const int*   b2a      = (const int*)d_in[19];
  const int*   b2revb   = (const int*)d_in[20];
  float* out = (float*)d_out;

  // ---- bump allocator over workspace ----
  char* wp = (char*)d_ws;
  auto take = [&](size_t bytes) -> void* {
    void* r = (void*)wp;
    wp += (bytes + 255) & ~(size_t)255;
    return r;
  };
  bf16* fatom_bf = (bf16*)take((size_t)MA * KAT * 2);
  bf16* fbond_bf = (bf16*)take((size_t)MB * KBD * 2);
  bf16* WiaT  = (bf16*)take((size_t)NPH * KAT * 2);
  bf16* WibT  = (bf16*)take((size_t)NPH * KBD * 2);
  bf16* Wh0T  = (bf16*)take((size_t)NPH * KH  * 2);
  bf16* Wh1T  = (bf16*)take((size_t)NPH * KH  * 2);
  bf16* WlrT  = (bf16*)take((size_t)NPH * KC  * 2);
  bf16* WoT   = (bf16*)take((size_t)NPH * KO  * 2);
  bf16* WihfT = (bf16*)take((size_t)NPG * KH  * 2);
  bf16* WhhfT = (bf16*)take((size_t)NPG * KH  * 2);
  bf16* WihbT = (bf16*)take((size_t)NPG * KH  * 2);
  bf16* WhhbT = (bf16*)take((size_t)NPG * KH  * 2);
  bf16* mb_bf   = (bf16*)take((size_t)MB * KH * 2);
  bf16* Acat_bf = (bf16*)take((size_t)MA * KC * 2);
  bf16* X_bf    = (bf16*)take((size_t)MX * KH * 2);
  bf16* gru_bf  = (bf16*)take((size_t)MA * KO * 2);
  bf16* hFbf    = (bf16*)take((size_t)NMOL * KH * 2);
  bf16* hBbf    = (bf16*)take((size_t)NMOL * KH * 2);
  float* input_atom = (float*)take((size_t)NATOMS * HDIM * 4);
  float* input_bond = (float*)take((size_t)NBONDS * HDIM * 4);
  float* matom      = (float*)take((size_t)NATOMS * HDIM * 4);
  float* mbond      = (float*)take((size_t)NBONDS * HDIM * 4);
  float* agg3       = (float*)take((size_t)NATOMS * HDIM * 4);
  float* agg2       = (float*)take((size_t)NATOMS * HDIM * 4);
  float* GIf        = (float*)take((size_t)MX * 3 * HDIM * 4);
  float* GIb        = (float*)take((size_t)MX * 3 * HDIM * 4);
  float* ghF        = (float*)take((size_t)NMOL * 3 * HDIM * 4);
  float* ghB        = (float*)take((size_t)NMOL * 3 * HDIM * 4);
  float* hF         = (float*)take((size_t)NMOL * HDIM * 4);
  float* hB         = (float*)take((size_t)NMOL * HDIM * 4);
  float* h0         = (float*)take((size_t)NMOL * HDIM * 4);
  float* AH         = (float*)take((size_t)NATOMS * HDIM * 4);

  auto blocks = [](long n) { return (unsigned)((n + TPB - 1) / TPB); };
  auto gemm = [&](const bf16* Ab, const bf16* Bt, const float* bias,
                  const float* resid, float* C, int M, int Mpad, int N, int Kp,
                  int mode) {
    dim3 g((unsigned)(Mpad / 256), (unsigned)((N + 63) / 64));
    k_wmma_gemm<<<g, 256, 0, stream>>>(Ab, Bt, bias, resid, C, M, N, Kp, N, mode);
  };

  // ---- conversions (weights transposed+padded, activations padded) ----
  {
    long n;
    n = (long)MA * KAT; k_conv_act<<<blocks(n), TPB, 0, stream>>>(f_atoms, fatom_bf, NATOMS, 133, KAT, n);
    n = (long)MB * KBD; k_conv_act<<<blocks(n), TPB, 0, stream>>>(f_bonds, fbond_bf, NBONDS, 147, KBD, n);
    n = (long)NPH * KAT; k_conv_w<<<blocks(n), TPB, 0, stream>>>(W_i_atom, WiaT, 133, HDIM, KAT, 0, n);
    n = (long)NPH * KBD; k_conv_w<<<blocks(n), TPB, 0, stream>>>(W_i_bond, WibT, 147, HDIM, KBD, 0, n);
    n = (long)NPH * KH;  k_conv_w<<<blocks(n), TPB, 0, stream>>>(W_h_0, Wh0T, HDIM, HDIM, KH, 0, n);
    k_conv_w<<<blocks(n), TPB, 0, stream>>>(W_h_1, Wh1T, HDIM, HDIM, KH, 0, n);
    n = (long)NPH * KC;  k_conv_w<<<blocks(n), TPB, 0, stream>>>(W_lr, WlrT, 3*HDIM, HDIM, KC, 0, n);
    n = (long)NPH * KO;  k_conv_w<<<blocks(n), TPB, 0, stream>>>(W_o, WoT, 2*HDIM, HDIM, KO, 0, n);
    n = (long)NPG * KH;  // GRU weights used as x @ W.T  ->  B = W.T (mode 1)
    k_conv_w<<<blocks(n), TPB, 0, stream>>>(Wih_f, WihfT, HDIM, 3*HDIM, KH, 1, n);
    k_conv_w<<<blocks(n), TPB, 0, stream>>>(Whh_f, WhhfT, HDIM, 3*HDIM, KH, 1, n);
    k_conv_w<<<blocks(n), TPB, 0, stream>>>(Wih_b, WihbT, HDIM, 3*HDIM, KH, 1, n);
    k_conv_w<<<blocks(n), TPB, 0, stream>>>(Whh_b, WhhbT, HDIM, 3*HDIM, KH, 1, n);
    // zero-fill padded bf16 staging buffers (pads must be 0 for the GEMMs)
    n = (long)MB * KH;  k_zero_bf<<<blocks((n + 7) / 8), TPB, 0, stream>>>(mb_bf, n);
    n = (long)MA * KC;  k_zero_bf<<<blocks((n + 7) / 8), TPB, 0, stream>>>(Acat_bf, n);
    n = (long)MX * KH;  k_zero_bf<<<blocks((n + 7) / 8), TPB, 0, stream>>>(X_bf, n);
    n = (long)MA * KO;  k_zero_bf<<<blocks((n + 7) / 8), TPB, 0, stream>>>(gru_bf, n);
    n = (long)NMOL * KH; k_zero_bf<<<blocks((n + 7) / 8), TPB, 0, stream>>>(hFbf, n);
    k_zero_bf<<<blocks((n + 7) / 8), TPB, 0, stream>>>(hBbf, n);
  }

  // ---- input projections ----
  gemm(fatom_bf, WiaT, nullptr, nullptr, input_atom, NATOMS, MA, HDIM, KAT, 1);
  gemm(fbond_bf, WibT, nullptr, nullptr, input_bond, NBONDS, MB, HDIM, KBD, 1);

  // ---- message passing step 1 (message_bond = input_bond) ----
  long natomsH = (long)NATOMS * HDIM, nbondsH = (long)NBONDS * HDIM;
  k_aggregate<<<blocks(natomsH), TPB, 0, stream>>>(input_bond, a2b, input_atom, matom);
  k_bond_pre<<<blocks(nbondsH), TPB, 0, stream>>>(matom, input_bond, b2a, b2revb, mb_bf);
  gemm(mb_bf, Wh0T, nullptr, input_bond, mbond, NBONDS, MB, HDIM, KH, 1 | 4);

  // ---- message passing step 2 ----
  k_aggregate<<<blocks(natomsH), TPB, 0, stream>>>(mbond, a2b, matom, matom);
  k_bond_pre<<<blocks(nbondsH), TPB, 0, stream>>>(matom, mbond, b2a, b2revb, mb_bf);
  gemm(mb_bf, Wh1T, nullptr, input_bond, mbond, NBONDS, MB, HDIM, KH, 1 | 4);

  // ---- final aggregate + concat + W_lr ----
  k_aggregate<<<blocks(natomsH), TPB, 0, stream>>>(mbond, a2b, nullptr, agg3);
  k_concat<<<blocks(natomsH), TPB, 0, stream>>>(agg3, matom, input_atom, Acat_bf);
  gemm(Acat_bf, WlrT, nullptr, nullptr, agg2, NATOMS, MA, HDIM, KC, 0);

  // ---- GRU setup ----
  k_msgx<<<blocks((long)MX * HDIM), TPB, 0, stream>>>(agg2, gru_bias, X_bf);
  long bh = (long)NMOL * HDIM;
  k_h0<<<blocks(bh), TPB, 0, stream>>>(agg2, h0);
  k_hinit<<<blocks(bh), TPB, 0, stream>>>(h0, hF, hB, hFbf, hBbf);
  k_row0<<<1, 320, 0, stream>>>(agg2, gru_bias, gru_bf);
  // GI = X @ Wih.T + bih for all timesteps, both directions
  gemm(X_bf, WihfT, bih_f, nullptr, GIf, MX, MX, 3 * HDIM, KH, 2);
  gemm(X_bf, WihbT, bih_b, nullptr, GIb, MX, MX, 3 * HDIM, KH, 2);

  // ---- forward GRU scan ----
  for (int t = 0; t < APM; ++t) {
    gemm(hFbf, WhhfT, bhh_f, nullptr, ghF, NMOL, NMOL, 3 * HDIM, KH, 2);
    k_gate<<<blocks(bh), TPB, 0, stream>>>(GIf, ghF, hF, hFbf, gru_bf, t, 0);
  }
  // ---- backward GRU scan ----
  for (int s = 0; s < APM; ++s) {
    int t = APM - 1 - s;
    gemm(hBbf, WhhbT, bhh_b, nullptr, ghB, NMOL, NMOL, 3 * HDIM, KH, 2);
    k_gate<<<blocks(bh), TPB, 0, stream>>>(GIb, ghB, hB, hBbf, gru_bf, t, HDIM);
  }

  // ---- output projection + mean pooling ----
  gemm(gru_bf, WoT, b_o, nullptr, AH, NATOMS, MA, HDIM, KO, 1 | 2);
  k_mean<<<blocks(bh), TPB, 0, stream>>>(AH, out);
}
